// RecurrentEmbeddingModule_38749194945029
// MI455X (gfx1250) — matrix operand order repaired and checked by reading, not compiled, not run on hardware
//
#include <hip/hip_runtime.h>

// ---------------------------------------------------------------------------
// Types / helpers
// ---------------------------------------------------------------------------
typedef __attribute__((ext_vector_type(16))) _Float16 v16h;
typedef __attribute__((ext_vector_type(8)))  float    v8f;

union AFrag { v16h v; _Float16 e[16]; };
union CFrag { v8f  v; float    f[8];  };

__device__ __forceinline__ float sigmoidf_(float x) { return 1.0f / (1.0f + __expf(-x)); }

// LDS offset of a (generic) pointer to __shared__ memory: flat LDS addresses
// carry the wave-relative LDS offset in addr[31:0] (aperture mapping).
__device__ __forceinline__ unsigned int lds_off_(const void* p) {
    return (unsigned int)(unsigned long long)p;
}

// Issue one 16-byte async global->LDS copy (tracked by ASYNCcnt).
__device__ __forceinline__ void async_copy_b128(unsigned int ldsOff, const void* gptr) {
    unsigned long long ga = (unsigned long long)gptr;
    asm volatile("global_load_async_to_lds_b128 %0, %1, off"
                 :: "v"(ldsOff), "v"(ga) : "memory");
}

__device__ __forceinline__ void async_wait0_() {
    asm volatile("s_wait_asynccnt 0" ::: "memory");
}

// ---------------------------------------------------------------------------
// Elementwise / conversion kernels
// ---------------------------------------------------------------------------
__global__ void k_f32_to_f16(const float* __restrict__ s, _Float16* __restrict__ d, int n) {
    int i = blockIdx.x * blockDim.x + threadIdx.x;
    if (i < n) d[i] = (_Float16)s[i];
}

// Pack a row-major f32 weight [N][K] into f16 [N][Kpad], zero padded.
__global__ void k_pack_w(const float* __restrict__ s, _Float16* __restrict__ d,
                         int K, int Kpad, int total) {
    int i = blockIdx.x * blockDim.x + threadIdx.x;
    if (i >= total) return;
    int n = i / Kpad, k = i - n * Kpad;
    d[i] = (k < K) ? (_Float16)s[(long)n * K + k] : (_Float16)0.0f;
}

// ---------------------------------------------------------------------------
// Implicit-GEMM conv (KSxKS, stride 1, pad KS/2) via WMMA f16->f32.
// One wave computes a 16(M=pixels) x 16(N=out channels) tile of one image.
// B (weight) tile is staged into LDS with async b128 copies each K-step.
// src: f16 [img][Cin][H][W]; wt: f16 [Cout][Kpad] (k = ci*KS*KS + ky*KS + kx)
// dst: f16 [img][Cout][H][W]; optional f32 bias[Cout], optional f16 skip.
// ---------------------------------------------------------------------------
template <int KS>
__global__ __launch_bounds__(32)
void k_conv_wmma(const _Float16* __restrict__ src, const _Float16* __restrict__ wt,
                 const float* __restrict__ bias, const _Float16* __restrict__ skip,
                 _Float16* __restrict__ dst,
                 int Cin, int H, int W, int K, int Kpad, int Cout) {
    __shared__ _Float16 sB[16 * 32];          // weight tile: 16 N x 32 K
    const int lane = threadIdx.x;
    const int half = lane >> 4;
    const int l15  = lane & 15;
    const int mtile = blockIdx.x, ntile = blockIdx.y, img = blockIdx.z;
    const int HW = H * W;
    const int m  = mtile * 16 + l15;          // output pixel (A row)
    const int py = m / W, px = m - py * W;
    const int n  = ntile * 16 + l15;          // output channel (B col)
    const long srcBase = (long)img * Cin * HW;

    CFrag c; c.v = {};
    for (int kk = 0; kk < Kpad; kk += 32) {
        // ---- async-stage 16x32 f16 weight tile into LDS (2 x b128 per lane)
#pragma unroll
        for (int i = 0; i < 2; ++i) {
            int e   = lane + i * 32;          // 0..63
            int row = e >> 2;                 // 0..15 (N within tile)
            int col = (e & 3) * 8;            // 0,8,16,24 (K within tile)
            const _Float16* g = wt + (long)(ntile * 16 + row) * Kpad + kk + col;
            async_copy_b128(lds_off_(&sB[row * 32 + col]), g);
        }
        async_wait0_();                        // single wave: own cnt suffices

        AFrag a, b;
#pragma unroll
        for (int e = 0; e < 16; ++e) {
            // A-matrix 16x32 f16 layout: lane(M)=l15; K = (e<8 ? e : e+8) + half*8
            int k = kk + (e < 8 ? e : e + 8) + half * 8;
            _Float16 av = (_Float16)0.0f;
            if (k < K) {
                if (KS == 3) {
                    int ci = k / 9, r = k - ci * 9;
                    int ky = r / 3, kx = r - ky * 3;
                    int iy = py + ky - 1, ix = px + kx - 1;
                    if ((unsigned)iy < (unsigned)H && (unsigned)ix < (unsigned)W)
                        av = src[srcBase + (long)ci * HW + (long)iy * W + ix];
                } else {  // 1x1
                    av = src[srcBase + (long)k * HW + m];
                }
            }
            a.e[e] = av;
            // B-matrix 32x16 f16 layout: lane(N)=l15; K = (e<8 ? e : e+8) + half*8
            b.e[e] = sB[l15 * 32 + (e < 8 ? e : e + 8) + half * 8];
        }
        c.v = __builtin_amdgcn_wmma_f32_16x16x32_f16(false, a.v, false, b.v,
                                                     (short)0, c.v, false, false);
    }

    const float bv = bias ? bias[n] : 0.0f;
    const long dbase = (long)img * Cout * HW + (long)n * HW;
#pragma unroll
    for (int r = 0; r < 8; ++r) {
        int mm = mtile * 16 + r + half * 8;   // C/D layout: M = r + half*8, N = l15
        float acc = c.f[r] + bv;
        if (skip) acc += (float)skip[dbase + mm];
        dst[dbase + mm] = (_Float16)acc;
    }
}

// ---------------------------------------------------------------------------
// Generic WMMA GEMM: out[M][N] = epi(A[M][K](f16, lda) * Wt[N][K]^T + bias)
// Block = 4 waves = 64(M) x 16(N) tile; A tile async-staged into LDS.
// EPI: 0 = none, 1 = SiLU, 2 = softplus, 3 = residual add (resid f32 [M][N])
// Grid: (M/64, N/16), 128 threads. Requires K % 32 == 0, lda % 8 == 0.
// ---------------------------------------------------------------------------
template <int EPI>
__global__ __launch_bounds__(128)
void k_gemm_wmma(const _Float16* __restrict__ A, int lda,
                 const _Float16* __restrict__ Wt,
                 const float* __restrict__ bias, const float* __restrict__ resid,
                 float* __restrict__ out, int K, int N) {
    __shared__ _Float16 sA[64 * 32];          // A tile: 64 M x 32 K
    const int tid  = threadIdx.x;
    const int wave = tid >> 5;
    const int lane = tid & 31;
    const int half = lane >> 4;
    const int l15  = lane & 15;
    const int mblk = blockIdx.x, ntile = blockIdx.y;
    const int n = ntile * 16 + l15;
    const int mlocal = wave * 16 + l15;
    const _Float16* wrow = Wt + (long)n * K;

    CFrag c; c.v = {};
    for (int kk = 0; kk < K; kk += 32) {
        __syncthreads();                      // previous-iter readers done
        // ---- async-stage 64x32 f16 A tile: 256 x b128 over 128 threads
#pragma unroll
        for (int i = 0; i < 2; ++i) {
            int e   = tid + i * 128;          // 0..255
            int row = e >> 2;                 // 0..63
            int col = (e & 3) * 8;            // 0,8,16,24
            const _Float16* g = A + (long)(mblk * 64 + row) * lda + kk + col;
            async_copy_b128(lds_off_(&sA[row * 32 + col]), g);
        }
        async_wait0_();
        __builtin_amdgcn_s_wait_tensorcnt(0);
        __syncthreads();                      // all waves' tiles visible

        __builtin_prefetch(wrow + kk + 128, 0, 1);   // global_prefetch_b8
        AFrag a, b;
#pragma unroll
        for (int e = 0; e < 16; ++e) {
            a.e[e] = sA[mlocal * 32 + (e < 8 ? e : e + 8) + half * 8];
            b.e[e] = wrow[kk + (e < 8 ? e : e + 8) + half * 8];
        }
        c.v = __builtin_amdgcn_wmma_f32_16x16x32_f16(false, a.v, false, b.v,
                                                     (short)0, c.v, false, false);
    }

    const float bv = bias ? bias[n] : 0.0f;
#pragma unroll
    for (int r = 0; r < 8; ++r) {
        int mm = mblk * 64 + wave * 16 + r + half * 8;
        long idx = (long)mm * N + n;
        float acc = c.f[r] + bv;
        if (EPI == 1) acc = acc * sigmoidf_(acc);                      // SiLU
        if (EPI == 2) acc = (acc > 20.0f) ? acc : log1pf(expf(acc));   // softplus
        if (EPI == 3) acc += resid[idx];                               // residual
        out[idx] = acc;
    }
}

// ---------------------------------------------------------------------------
// GroupNorm(1 group over C*H*W) + SiLU, per image (one block per image)
// ---------------------------------------------------------------------------
__global__ void k_gn_silu(const _Float16* __restrict__ x, const float* __restrict__ gs,
                          const float* __restrict__ gb, _Float16* __restrict__ h,
                          int C, int HW) {
    __shared__ float s1[256], s2[256];
    const int img = blockIdx.x, tid = threadIdx.x;
    const int n = C * HW;
    const long base = (long)img * n;
    float sum = 0.0f, sq = 0.0f;
    for (int i = tid; i < n; i += 256) {
        float v = (float)x[base + i];
        sum += v; sq += v * v;
    }
    s1[tid] = sum; s2[tid] = sq;
    __syncthreads();
    for (int off = 128; off > 0; off >>= 1) {
        if (tid < off) { s1[tid] += s1[tid + off]; s2[tid] += s2[tid + off]; }
        __syncthreads();
    }
    const float mean = s1[0] / n;
    const float var  = s2[0] / n - mean * mean;
    const float rs   = rsqrtf(var + 1e-5f);
    for (int i = tid; i < n; i += 256) {
        int c = i / HW;
        float v = ((float)x[base + i] - mean) * rs * gs[c] + gb[c];
        h[base + i] = (_Float16)(v * sigmoidf_(v));
    }
}

// 2x2 max pool, f16 -> f16
__global__ void k_maxpool(const _Float16* __restrict__ s, _Float16* __restrict__ d,
                          int C, int H, int W, int total) {
    int i = blockIdx.x * blockDim.x + threadIdx.x;
    if (i >= total) return;
    int ow = W >> 1, oh = H >> 1;
    int ox = i % ow; int t = i / ow;
    int oy = t % oh; t /= oh;
    int c  = t % C;  int img = t / C;
    long b = ((long)img * C + c) * H * W + (long)(oy * 2) * W + ox * 2;
    float m = fmaxf(fmaxf((float)s[b], (float)s[b + 1]),
                    fmaxf((float)s[b + W], (float)s[b + W + 1]));
    d[i] = (_Float16)m;
}

// Copy pooled features [img][4096] into feat rows [img][4352] (first 4096 cols)
__global__ void k_featcopy(const _Float16* __restrict__ s, _Float16* __restrict__ d, int total) {
    int i = blockIdx.x * blockDim.x + threadIdx.x;
    if (i >= total) return;
    int img = i >> 12, j = i & 4095;
    d[(long)img * 4352 + j] = s[i];
}

// feat[tok][4096+j] = act_emb[act[tok]][j]
__global__ void k_actfill(const float* __restrict__ emb, const int* __restrict__ act,
                          _Float16* __restrict__ feat) {
    int i = blockIdx.x * blockDim.x + threadIdx.x;
    if (i >= 512 * 256) return;
    int j = i & 255, tok = i >> 8;
    feat[(long)tok * 4352 + 4096 + j] = (_Float16)emb[(long)act[tok] * 256 + j];
}

// RMSNorm per row (one block per row)
template <typename OutT>
__global__ void k_rmsnorm(const float* __restrict__ x, const float* __restrict__ w,
                          OutT* __restrict__ o, int D) {
    __shared__ float s1[256];
    const int row = blockIdx.x, tid = threadIdx.x;
    const float* xr = x + (long)row * D;
    float sq = 0.0f;
    for (int i = tid; i < D; i += 256) { float v = xr[i]; sq += v * v; }
    s1[tid] = sq;
    __syncthreads();
    for (int off = 128; off > 0; off >>= 1) {
        if (tid < off) s1[tid] += s1[tid + off];
        __syncthreads();
    }
    const float scale = rsqrtf(s1[0] / D + 1e-5f);
    for (int i = tid; i < D; i += 256)
        o[(long)row * D + i] = (OutT)(xr[i] * scale * w[i]);
}

// Causal depthwise conv1d (D_CONV=4) + SiLU over xz[:, :1024]; out f16 [bt][1024]
__global__ void k_conv1d_silu(const float* __restrict__ xz, const float* __restrict__ w,
                              const float* __restrict__ b, _Float16* __restrict__ xc) {
    int i = blockIdx.x * blockDim.x + threadIdx.x;
    if (i >= 512 * 1024) return;
    int d = i & 1023, bt = i >> 10, t = bt & 63;
    float acc = b[d];
#pragma unroll
    for (int j = 0; j < 4; ++j) {
        int tt = t - 3 + j;
        if (tt >= 0) acc += xz[(long)(bt + tt - t) * 2048 + d] * w[d * 4 + j];
    }
    xc[i] = (_Float16)(acc * sigmoidf_(acc));
}

// Selective scan: one thread per (b, d_inner) channel, T=64, N=16 states
__global__ void k_scan(const float* __restrict__ delta, const float* __restrict__ dbc,
                       const _Float16* __restrict__ xc, const float* __restrict__ xz,
                       const float* __restrict__ Alog, const float* __restrict__ Dv,
                       _Float16* __restrict__ y) {
    int i = blockIdx.x * blockDim.x + threadIdx.x;
    if (i >= 8 * 1024) return;
    int d = i & 1023, b = i >> 10;
    float Ar[16], h[16];
#pragma unroll
    for (int n = 0; n < 16; ++n) { Ar[n] = -__expf(Alog[(long)d * 16 + n]); h[n] = 0.0f; }
    const float Dd = Dv[d];
    for (int t = 0; t < 64; ++t) {
        int bt = b * 64 + t;
        float dl = delta[(long)bt * 1024 + d];
        float xv = (float)xc[(long)bt * 1024 + d];
        const float* row = dbc + (long)bt * 64;
        float acc = 0.0f;
#pragma unroll
        for (int n = 0; n < 16; ++n) {
            float dA = __expf(dl * Ar[n]);
            h[n] = dA * h[n] + dl * row[32 + n] * xv;
            acc += h[n] * row[48 + n];
        }
        acc += Dd * xv;
        float z = xz[(long)bt * 2048 + 1024 + d];
        acc *= z * sigmoidf_(z);
        y[(long)bt * 1024 + d] = (_Float16)acc;
    }
}

// ---------------------------------------------------------------------------
// Host orchestration
// ---------------------------------------------------------------------------
extern "C" void kernel_launch(void* const* d_in, const int* in_sizes, int n_in,
                              void* d_out, int out_size, void* d_ws, size_t ws_size,
                              hipStream_t stream) {
    auto F = [&](int i) { return (const float*)d_in[i]; };
    const float* obs = F(0);
    const int*   act = (const int*)d_in[1];
    int pi = 2;
    const float* conv0_w = F(pi++); const float* conv0_b = F(pi++);
    const float *gs[4], *gb[4], *cw[4], *cb[4], *pw[4], *pb[4];
    for (int i = 0; i < 4; ++i) {
        gs[i] = F(pi++); gb[i] = F(pi++); cw[i] = F(pi++); cb[i] = F(pi++);
        if (i > 0) { pw[i] = F(pi++); pb[i] = F(pi++); } else { pw[i] = nullptr; pb[i] = nullptr; }
    }
    const float* act_emb = F(pi++);
    const float* proj_w  = F(pi++);
    const float* proj_b  = F(pi++);
    const float *nw[3], *ipw[3], *c1w[3], *c1b[3], *xpw[3], *dtw[3], *dtb[3], *alog[3], *Dvp[3], *opw[3];
    for (int l = 0; l < 3; ++l) {
        nw[l] = F(pi++); ipw[l] = F(pi++); c1w[l] = F(pi++); c1b[l] = F(pi++);
        xpw[l] = F(pi++); dtw[l] = F(pi++); dtb[l] = F(pi++); alog[l] = F(pi++);
        Dvp[l] = F(pi++); opw[l] = F(pi++);
    }
    const float* normf = F(pi++);

    // ---- workspace carve ----
    char* wp = (char*)d_ws;
    auto allocB = [&](size_t bytes) -> char* {
        char* r = wp; wp += (bytes + 255) & ~(size_t)255; return r;
    };
    auto allocH = [&](size_t n) { return (_Float16*)allocB(n * 2); };
    auto allocF = [&](size_t n) { return (float*)allocB(n * 4); };

    _Float16* arenaIn = allocH(786432);      // 64 imgs * 3*64*64
    _Float16* arenaX  = allocH(8388608);     // 64 imgs * 32*64*64 (max stage)
    _Float16* arenaH  = allocH(8388608);
    _Float16* arenaY  = allocH(8388608);
    _Float16* arenaS  = allocH(4194304);     // proj skip (max 64 imgs * 64*32*32)
    _Float16* feat_h  = allocH((size_t)512 * 4352);
    float*    xbuf    = allocF((size_t)512 * 512);
    _Float16* xr_h    = allocH((size_t)512 * 512);
    float*    xz      = allocF((size_t)512 * 2048);
    _Float16* xc_h    = allocH((size_t)512 * 1024);
    float*    dbc     = allocF((size_t)512 * 64);
    _Float16* dbc_h   = allocH((size_t)512 * 64);
    float*    deltab  = allocF((size_t)512 * 1024);
    _Float16* ybuf_h  = allocH((size_t)512 * 1024);

    // packed f16 weights
    _Float16* pk_conv0 = allocH(32 * 32);
    const int CH[4] = {32, 64, 128, 256};
    const int cin_l[4] = {32, 32, 64, 128};
    _Float16 *pk_rconv[4], *pk_rproj[4] = {nullptr, nullptr, nullptr, nullptr};
    for (int i = 0; i < 4; ++i) {
        pk_rconv[i] = allocH((size_t)CH[i] * cin_l[i] * 9);
        if (pw[i]) pk_rproj[i] = allocH((size_t)CH[i] * cin_l[i]);
    }
    _Float16* pk_proj = allocH((size_t)512 * 4352);
    _Float16 *pk_in[3], *pk_xp[3], *pk_dt[3], *pk_op[3];
    for (int l = 0; l < 3; ++l) {
        pk_in[l] = allocH((size_t)2048 * 512);
        pk_xp[l] = allocH((size_t)64 * 1024);
        pk_dt[l] = allocH((size_t)1024 * 32);
        pk_op[l] = allocH((size_t)512 * 1024);
    }

    auto pack = [&](const float* src, _Float16* dst, int N, int K, int Kpad) {
        int tot = N * Kpad;
        k_pack_w<<<(tot + 255) / 256, 256, 0, stream>>>(src, dst, K, Kpad, tot);
    };

    // ---- pack all weights (every call; deterministic) ----
    pack(conv0_w, pk_conv0, 32, 27, 32);
    for (int i = 0; i < 4; ++i) {
        pack(cw[i], pk_rconv[i], CH[i], cin_l[i] * 9, cin_l[i] * 9);
        if (pw[i]) pack(pw[i], pk_rproj[i], CH[i], cin_l[i], cin_l[i]);
    }
    pack(proj_w, pk_proj, 512, 4352, 4352);
    for (int l = 0; l < 3; ++l) {
        pack(ipw[l], pk_in[l], 2048, 512, 512);
        pack(xpw[l], pk_xp[l], 64, 1024, 1024);
        pack(dtw[l], pk_dt[l], 1024, 32, 32);
        pack(opw[l], pk_op[l], 512, 1024, 1024);
    }

    // ---- encoder: 8 chunks of 64 images ----
    for (int ch = 0; ch < 8; ++ch) {
        const float* obsC = obs + (size_t)ch * 64 * 3 * 4096;
        int nIn = 64 * 3 * 4096;
        k_f32_to_f16<<<(nIn + 255) / 256, 256, 0, stream>>>(obsC, arenaIn, nIn);

        // conv0: 3 -> 32, 64x64, K=27 padded to 32
        {
            dim3 g(4096 / 16, 32 / 16, 64);
            k_conv_wmma<3><<<g, 32, 0, stream>>>(arenaIn, pk_conv0, conv0_b, nullptr,
                                                 arenaX, 3, 64, 64, 27, 32, 32);
        }

        int H = 64, W = 64, cin = 32;
        for (int i = 0; i < 4; ++i) {
            int cout = CH[i], HW = H * W;
            k_gn_silu<<<64, 256, 0, stream>>>(arenaX, gs[i], gb[i], arenaH, cin, HW);

            const _Float16* skip;
            if (pw[i]) {
                dim3 gp(HW / 16, cout / 16, 64);
                k_conv_wmma<1><<<gp, 32, 0, stream>>>(arenaX, pk_rproj[i], pb[i], nullptr,
                                                      arenaS, cin, H, W, cin, cin, cout);
                skip = arenaS;
            } else {
                skip = arenaX;
            }
            dim3 gc(HW / 16, cout / 16, 64);
            k_conv_wmma<3><<<gc, 32, 0, stream>>>(arenaH, pk_rconv[i], cb[i], skip,
                                                  arenaY, cin, H, W, cin * 9, cin * 9, cout);

            int tot = 64 * cout * (H / 2) * (W / 2);
            k_maxpool<<<(tot + 255) / 256, 256, 0, stream>>>(arenaY, arenaX, cout, H, W, tot);
            H >>= 1; W >>= 1; cin = cout;
        }
        // arenaX now [64][256*4*4 = 4096]
        int totf = 64 * 4096;
        k_featcopy<<<(totf + 255) / 256, 256, 0, stream>>>(
            arenaX, feat_h + (size_t)ch * 64 * 4352, totf);
    }

    k_actfill<<<(512 * 256) / 256, 256, 0, stream>>>(act_emb, act, feat_h);

    // proj: silu(feat @ W^T + b), M=512 K=4352 N=512   (grid: M/64 x N/16)
    k_gemm_wmma<1><<<dim3(8, 32), 128, 0, stream>>>(feat_h, 4352, pk_proj, proj_b,
                                                    nullptr, xbuf, 4352, 512);

    // ---- Mamba layers ----
    for (int l = 0; l < 3; ++l) {
        k_rmsnorm<_Float16><<<512, 256, 0, stream>>>(xbuf, nw[l], xr_h, 512);
        k_gemm_wmma<0><<<dim3(8, 128), 128, 0, stream>>>(xr_h, 512, pk_in[l], nullptr,
                                                         nullptr, xz, 512, 2048);
        k_conv1d_silu<<<(512 * 1024) / 256, 256, 0, stream>>>(xz, c1w[l], c1b[l], xc_h);
        k_gemm_wmma<0><<<dim3(8, 4), 128, 0, stream>>>(xc_h, 1024, pk_xp[l], nullptr,
                                                       nullptr, dbc, 1024, 64);
        k_f32_to_f16<<<(512 * 64) / 256, 256, 0, stream>>>(dbc, dbc_h, 512 * 64);
        k_gemm_wmma<2><<<dim3(8, 64), 128, 0, stream>>>(dbc_h, 64, pk_dt[l], dtb[l],
                                                        nullptr, deltab, 32, 1024);
        k_scan<<<8192 / 256, 256, 0, stream>>>(deltab, dbc, xc_h, xz, alog[l], Dvp[l], ybuf_h);
        k_gemm_wmma<3><<<dim3(8, 32), 128, 0, stream>>>(ybuf_h, 1024, pk_op[l], nullptr,
                                                        xbuf, xbuf, 1024, 512);
    }

    k_rmsnorm<float><<<512, 256, 0, stream>>>(xbuf, normf, (float*)d_out, 512);
    (void)in_sizes; (void)n_in; (void)out_size; (void)ws_size;
}
